// AttentionLayer_26053271617571
// MI455X (gfx1250) — matrix operand order repaired and checked by reading, not compiled, not run on hardware
//
#include <hip/hip_runtime.h>

// ---- problem constants (match reference) ----
#define B_TOT 65536
#define NN    20      // neighbors
#define CC    128     // channels
#define HH    4       // heads
#define HDD   32      // head dim
#define BT    4       // batch elems per workgroup
#define ROWS  (BT*NN) // 80 = 5 * 16  -> zero M-tile padding
#define MT    (ROWS/16)

typedef __attribute__((ext_vector_type(16))) _Float16 v16h;
typedef __attribute__((ext_vector_type(8)))  float    v8f;
typedef __attribute__((ext_vector_type(4)))  float    f32x4;
typedef __attribute__((ext_vector_type(4)))  unsigned u32x4;

union F16x16 { v16h v; u32x4 q[2]; };
union H8     { u32x4 u; _Float16 h[8]; };

#if __has_builtin(__builtin_amdgcn_global_load_async_to_lds_b128) && \
    __has_builtin(__builtin_amdgcn_s_wait_asynccnt)
#define USE_ASYNC_LDS 1
typedef int v4i __attribute__((vector_size(16)));       // matches builtin proto
typedef __attribute__((address_space(1))) v4i* gas_v4i; // global (AS1) int4*
typedef __attribute__((address_space(3))) v4i* las_v4i; // LDS (AS3) int4*
static __device__ __forceinline__ gas_v4i as_global_v4i(const void* p) {
  return (gas_v4i)(uintptr_t)p;                // AS1: full 64-bit address
}
static __device__ __forceinline__ las_v4i as_lds_v4i(void* p) {
  return (las_v4i)(unsigned)(uintptr_t)p;      // AS3: low 32 bits = LDS offset
}
#else
#define USE_ASYNC_LDS 0
#endif

// One-shot fp32 -> fp16 weight conversion into workspace.
// Layout: [Wq | Wk | Wv | Wo], each 128x128 row-major fp16 (32 KB each).
__global__ void convert_w_kernel(const float* __restrict__ Wq, const float* __restrict__ Wk,
                                 const float* __restrict__ Wv, const float* __restrict__ Wo,
                                 _Float16* __restrict__ w16) {
  int i = blockIdx.x * blockDim.x + threadIdx.x;   // 0..65535
  int m = i >> 14, r = i & 16383;
  float v = (m == 0) ? Wq[r] : (m == 1) ? Wk[r] : (m == 2) ? Wv[r] : Wo[r];
  w16[i] = (_Float16)v;
}

__global__ __launch_bounds__(256) void attn_fused_kernel(
    const float* __restrict__ x, const float* __restrict__ nbr,
    const _Float16* __restrict__ w16,
    const float* __restrict__ bq, const float* __restrict__ bk,
    const float* __restrict__ bv, const float* __restrict__ bo,
    const float* __restrict__ Wse1, const float* __restrict__ Wse2,
    const float* __restrict__ Wg, const float* __restrict__ bg,
    float* __restrict__ out)
{
  __shared__ _Float16 sNbr[ROWS][CC];      // 20 KB  neighbors fp16
  __shared__ _Float16 sX[16][CC];          //  4 KB  x fp16 (rows 4..15 zero)
  __shared__ float    sXf[BT][CC];         //  2 KB  x fp32 (residual)
  __shared__ _Float16 sKV[ROWS][CC];       // 20 KB  K then V
  __shared__ float    sQ[BT][CC];          //  2 KB
  __shared__ float    sScore[BT][HH][NN];  // 1.25 KB  scores -> attn
  __shared__ _Float16 sOutH[16][CC];       //  4 KB  attn-out (Wo input, permuted)
  __shared__ float    sO2[BT][CC];         //  2 KB  out after Wo (then after SE)
  __shared__ float    sSe[BT][32];
  __shared__ float    sGate[BT];

  const int tid   = threadIdx.x;
  const int lane  = tid & 31;
  const int wv_id = tid >> 5;              // 8 waves
  const long b0   = (long)blockIdx.x * BT;

  // Warm L2 for this wave's weight slice (global_prefetch_b8).
  __builtin_prefetch(w16 + wv_id * 8192, 0, 3);

  // ---------- Stage 0: stage neighbors (fp16) + x (fp32/fp16), zero pads ----------
#if USE_ASYNC_LDS
  // Async DMA of the fp32 residual tile straight into LDS (ASYNCcnt path).
  if (tid < BT * CC / 4) {
    __builtin_amdgcn_global_load_async_to_lds_b128(
        as_global_v4i(x + b0 * CC + tid * 4), as_lds_v4i(&sXf[0][tid * 4]), 0, 0);
  }
#endif
  {
    const f32x4* g = (const f32x4*)(nbr + b0 * NN * CC);
    for (int it = tid; it < ROWS * CC / 8; it += 256) {
      f32x4 a = __builtin_nontemporal_load(g + it * 2);
      f32x4 b = __builtin_nontemporal_load(g + it * 2 + 1);
      H8 t;
      t.h[0] = (_Float16)a.x; t.h[1] = (_Float16)a.y; t.h[2] = (_Float16)a.z; t.h[3] = (_Float16)a.w;
      t.h[4] = (_Float16)b.x; t.h[5] = (_Float16)b.y; t.h[6] = (_Float16)b.z; t.h[7] = (_Float16)b.w;
      *(u32x4*)(&sNbr[0][0] + it * 8) = t.u;          // ds_store_b128
    }
    for (int it = tid; it < (16 - BT) * CC; it += 256) {
      (&sX[BT][0])[it]    = (_Float16)0.f;            // pad rows (only rows 0..3 used)
      (&sOutH[BT][0])[it] = (_Float16)0.f;
    }
  }
#if USE_ASYNC_LDS
  __builtin_amdgcn_s_wait_asynccnt(0);
  __syncthreads();                                    // sXf now visible block-wide
  for (int it = tid; it < BT * CC; it += 256)
    (&sX[0][0])[it] = (_Float16)sXf[0][it];
#else
  for (int it = tid; it < BT * CC; it += 256) {
    float v = __builtin_nontemporal_load(x + b0 * CC + it);
    sXf[0][it] = v;
    (&sX[0][0])[it] = (_Float16)v;
  }
#endif
  __syncthreads();

  const int c  = lane & 15;   // N column within tile / A row within tile
  const int kh = lane >> 4;   // K-half selector (ISA fragment layout)

  // Single-M-tile GEMM: D = A(16x128,lds) @ W^T + bias -> dst rows 0..BT-1 (fp32)
  auto gemm1 = [&](const _Float16 (*A)[CC], int wsel, const float* bias, float (*dst)[CC]) {
    const int j0 = wv_id * 16;
    v8f acc = {};
    #pragma unroll
    for (int kb = 0; kb < 4; ++kb) {
      const int k0 = kb * 32;
      F16x16 a, bf;
      a.q[0]  = *(const u32x4*)&A[c][k0 + 8 * kh];
      a.q[1]  = *(const u32x4*)&A[c][k0 + 16 + 8 * kh];
      const _Float16* wr = w16 + wsel * 16384 + (j0 + c) * CC + k0 + 8 * kh;
      bf.q[0] = *(const u32x4*)wr;
      bf.q[1] = *(const u32x4*)(wr + 16);
      acc = __builtin_amdgcn_wmma_f32_16x16x32_f16(false, a.v, false, bf.v,
                                                   (short)0, acc, false, false);
    }
    if (lane < 16) {                                   // M rows 0..3 live in lanes 0..15
      float bj = bias[j0 + lane];
      #pragma unroll
      for (int i = 0; i < BT; ++i) dst[i][j0 + lane] = acc[i] + bj;
    }
  };

  // Big projection: sKV = sNbr(80x128) @ W^T + bias, fp16 result.
  auto project = [&](int wsel, const float* bias) {
    const int j0 = wv_id * 16;
    F16x16 bf[4];                                      // hoist B frags, reuse over 5 M-tiles
    #pragma unroll
    for (int kb = 0; kb < 4; ++kb) {
      const _Float16* wr = w16 + wsel * 16384 + (j0 + c) * CC + kb * 32 + 8 * kh;
      bf[kb].q[0] = *(const u32x4*)wr;
      bf[kb].q[1] = *(const u32x4*)(wr + 16);
    }
    const float bj = bias[j0 + c];
    for (int mt = 0; mt < MT; ++mt) {
      const int m0 = mt * 16;
      v8f acc = {};
      #pragma unroll
      for (int kb = 0; kb < 4; ++kb) {
        F16x16 a;
        a.q[0] = *(const u32x4*)&sNbr[m0 + c][kb * 32 + 8 * kh];
        a.q[1] = *(const u32x4*)&sNbr[m0 + c][kb * 32 + 16 + 8 * kh];
        acc = __builtin_amdgcn_wmma_f32_16x16x32_f16(false, a.v, false, bf[kb].v,
                                                     (short)0, acc, false, false);
      }
      const int col = j0 + c, rb = m0 + 8 * kh;        // D: VGPR i -> M = i + 8*(lane/16)
      #pragma unroll
      for (int i = 0; i < 8; ++i)
        sKV[rb + i][col] = (_Float16)(acc[i] + bj);
    }
  };

  // ---------- Stage 1: Q ---------- / ---------- Stage 2: K ----------
  gemm1(sX, 0, bq, sQ);
  __syncthreads();
  project(1, bk);
  __syncthreads();

  // ---------- Stage 3: scores + softmax (tiny, VALU) ----------
  const float scale = 0.17677669529663689f;            // 1/sqrt(32)
  for (int idx = tid; idx < BT * HH * NN; idx += 256) {
    int b = idx / (HH * NN), h = (idx / NN) % HH, n = idx % NN;
    const float*    qv = &sQ[b][h * HDD];
    const _Float16* kv = &sKV[b * NN + n][h * HDD];
    float s = 0.f;
    #pragma unroll
    for (int d = 0; d < HDD; ++d) s += qv[d] * (float)kv[d];
    sScore[b][h][n] = s * scale;
  }
  __syncthreads();
  if (tid < BT * HH) {
    int b = tid / HH, h = tid % HH;
    float mx = -1e30f;
    for (int n = 0; n < NN; ++n) mx = fmaxf(mx, sScore[b][h][n]);
    float e[NN], sum = 0.f;
    for (int n = 0; n < NN; ++n) { e[n] = __expf(sScore[b][h][n] - mx); sum += e[n]; }
    float inv = 1.f / sum;
    for (int n = 0; n < NN; ++n) sScore[b][h][n] = e[n] * inv;
  }
  __syncthreads();

  // ---------- Stage 4: V (overwrites K buffer) ----------
  project(2, bv);
  __syncthreads();

  // ---------- Stage 5: attn @ V, write with (0,2,1)-permute: col = d*H + h ----------
  for (int idx = tid; idx < BT * CC; idx += 256) {
    int b = idx >> 7, j = idx & 127, h = j >> 5, d = j & 31;
    float acc = 0.f;
    #pragma unroll
    for (int n = 0; n < NN; ++n)
      acc += sScore[b][h][n] * (float)sKV[b * NN + n][j];
    sOutH[b][d * HH + h] = (_Float16)acc;
  }
  __syncthreads();

  // ---------- Stage 6: Wo ----------
  gemm1(sOutH, 3, bo, sO2);
  __syncthreads();

  // ---------- Stage 7: SE gate ----------
  if (tid < BT * 32) {
    int b = tid >> 5, r = tid & 31;
    const float* wr = Wse1 + r * CC;
    float s = 0.f;
    for (int k = 0; k < CC; ++k) s += sO2[b][k] * wr[k];
    sSe[b][r] = fmaxf(s, 0.f);
  }
  __syncthreads();
  float seval[2];
  #pragma unroll
  for (int t = 0; t < 2; ++t) {
    int idx = tid + t * 256;
    int b = idx >> 7, j = idx & 127;
    const float* wr = Wse2 + j * 32;
    float s = 0.f;
    #pragma unroll
    for (int k = 0; k < 32; ++k) s += sSe[b][k] * wr[k];
    seval[t] = sO2[b][j] * (1.f / (1.f + __expf(-s)));
  }
  __syncthreads();
  #pragma unroll
  for (int t = 0; t < 2; ++t) {
    int idx = tid + t * 256;
    sO2[idx >> 7][idx & 127] = seval[t];
  }
  __syncthreads();

  // ---------- Stage 8: residual gate (one wave per batch elem) ----------
  if (tid < BT * 32) {
    int b = wv_id;
    float p = 0.f;
    for (int ci = lane; ci < 2 * CC; ci += 32) {
      float v = (ci < CC) ? sXf[b][ci] : sO2[b][ci - CC];
      p += Wg[ci] * v;
    }
    #pragma unroll
    for (int off = 16; off > 0; off >>= 1) p += __shfl_xor(p, off, 32);
    if (lane == 0) sGate[b] = 1.f / (1.f + __expf(-(p + bg[0])));
  }
  __syncthreads();

  // ---------- Stage 9: blend + store ----------
  for (int idx = tid; idx < BT * CC; idx += 256) {
    int b = idx >> 7, j = idx & 127;
    float g = sGate[b];
    float o = g * sO2[b][j] + (1.f - g) * sXf[b][j];
    __builtin_nontemporal_store(o, out + (b0 + b) * CC + j);
  }
}

extern "C" void kernel_launch(void* const* d_in, const int* in_sizes, int n_in,
                              void* d_out, int out_size, void* d_ws, size_t ws_size,
                              hipStream_t stream) {
  const float* x    = (const float*)d_in[0];
  const float* nbr  = (const float*)d_in[1];
  const float* Wq   = (const float*)d_in[2];
  const float* bq   = (const float*)d_in[3];
  const float* Wk   = (const float*)d_in[4];
  const float* bk   = (const float*)d_in[5];
  const float* Wv   = (const float*)d_in[6];
  const float* bv   = (const float*)d_in[7];
  const float* Wo   = (const float*)d_in[8];
  const float* bo   = (const float*)d_in[9];
  const float* Wse1 = (const float*)d_in[10];
  const float* Wse2 = (const float*)d_in[11];
  const float* Wg   = (const float*)d_in[12];
  const float* bg   = (const float*)d_in[13];
  _Float16* w16 = (_Float16*)d_ws;   // 128 KB: Wq|Wk|Wv|Wo fp16

  convert_w_kernel<<<256, 256, 0, stream>>>(Wq, Wk, Wv, Wo, w16);
  attn_fused_kernel<<<B_TOT / BT, 256, 0, stream>>>(
      x, nbr, w16, bq, bk, bv, bo, Wse1, Wse2, Wg, bg, (float*)d_out);
}
